// LoraBF16Linear_13881334300847
// MI455X (gfx1250) — matrix operand order repaired and checked by reading, not compiled, not run on hardware
//
#include <hip/hip_runtime.h>
#include <hip/hip_bf16.h>

// ---------------------------------------------------------------------------
// LoRA linear  y = x @ W^T + b + s*(x@A^T)@B^T   fused as  y = x @ Weff^T + b
// with Weff = bf16(W + s * B @ A)  precomputed into d_ws.
// Main GEMM: M=16384, N=4096, K=4096, bf16 WMMA (v_wmma_f32_16x16x32_bf16).
// Block tile 256x128, wave tile 64x64 (16 WMMA / K-tile / wave).
// Weff tile streamed with GLOBAL_LOAD_ASYNC_TO_LDS_B128 (ASYNCcnt).
// ---------------------------------------------------------------------------

typedef __attribute__((ext_vector_type(16))) __bf16 v16bf;
typedef __attribute__((ext_vector_type(8)))  __bf16 v8bf;
typedef __attribute__((ext_vector_type(4)))  __bf16 v4bf;
typedef __attribute__((ext_vector_type(2)))  __bf16 v2bf;
typedef __attribute__((ext_vector_type(8)))  float  v8f;

static __device__ __forceinline__ __bf16 f2bf_rne(float f) {
    unsigned u = __builtin_bit_cast(unsigned, f);
    unsigned r = (u + 0x7FFFu + ((u >> 16) & 1u)) >> 16;
    unsigned short s = (unsigned short)r;
    return __builtin_bit_cast(__bf16, s);
}

// Packed f32x2 -> bf16x2 conversion; native v_cvt_pk_bf16_f32 when available.
#if defined(__AMDGCN__) && __has_builtin(__builtin_amdgcn_cvt_pk_bf16_f32)
#define HAVE_CVT_PK_BF16 1
#endif

static __device__ __forceinline__ v2bf pack2bf(float a, float b) {
#ifdef HAVE_CVT_PK_BF16
    return __builtin_amdgcn_cvt_pk_bf16_f32(a, b);
#else
    v2bf r; r[0] = f2bf_rne(a); r[1] = f2bf_rne(b); return r;
#endif
}

// ---------------------------------------------------------------------------
// Kernel 1: Weff[o][k] = bf16( W[o][k] + scaling * sum_j B[o][j]*A[j][k] )
// ---------------------------------------------------------------------------
__global__ void __launch_bounds__(256)
fuse_weights_kernel(const float* __restrict__ W,
                    const float* __restrict__ A,
                    const float* __restrict__ Bm,
                    __bf16* __restrict__ Weff,
                    int Dout, int Din, int r, float scaling)
{
    const int o = blockIdx.y;
    const int k = (blockIdx.x * blockDim.x + threadIdx.x) * 4;
    if (k >= Din || o >= Dout) return;

    float4 w = *(const float4*)(W + (size_t)o * Din + k);
    float a0 = w.x, a1 = w.y, a2 = w.z, a3 = w.w;
    for (int j = 0; j < r; ++j) {
        const float bj = scaling * Bm[(size_t)o * r + j];
        float4 a = *(const float4*)(A + (size_t)j * Din + k);
        a0 += bj * a.x; a1 += bj * a.y; a2 += bj * a.z; a3 += bj * a.w;
    }
    v2bf lo = pack2bf(a0, a1);
    v2bf hi = pack2bf(a2, a3);
    v4bf out;
    out[0] = lo[0]; out[1] = lo[1]; out[2] = hi[0]; out[3] = hi[1];
    *(v4bf*)(Weff + (size_t)o * Din + k) = out;
}

// ---------------------------------------------------------------------------
// Kernel 2: Out[m][n] = sum_k bf16(X[m][k]) * Weff[n][k] + bias[n]
// 256 threads = 8 waves (wave32) in 4(M)x2(N) grid; each wave owns 64x64:
// 4x4 tiles of 16x16 via v_wmma_f32_16x16x32_bf16 (16 WMMA per K-tile).
// KK > 0 -> K is a compile-time constant (imm-offset saddr addressing).
// ---------------------------------------------------------------------------
#define BM   256
#define BN   128
#define BK   32
#define LDSS 40   // bf16 elems per LDS row: 32 data + 8 pad (80B, 16B aligned)

template<int KK>
__global__ void __launch_bounds__(256)
lora_gemm_kernel(const float* __restrict__ X,
                 const __bf16* __restrict__ Weff,
                 const float* __restrict__ bias,
                 float* __restrict__ Out,
                 int M, int N, int Kparam)
{
    const int KD = (KK > 0) ? KK : Kparam;   // K dimension

    __shared__ __align__(16) __bf16 As[2][BM * LDSS];  // 2*256*40*2 = 40960 B
    __shared__ __align__(16) __bf16 Bs[2][BN * LDSS];  // 2*128*40*2 = 20480 B

    const int tid     = threadIdx.x;
    const int lane    = tid & 31;
    const int wave    = tid >> 5;
    const int wm      = wave & 3;        // 0..3 -> 64-row slice of M
    const int wn      = wave >> 2;       // 0..1 -> 64-col slice of N
    const int l16     = lane & 15;
    const int halfSel = lane >> 4;       // 0 for lanes 0-15, 1 for 16-31

    const int mBlock = blockIdx.y * BM;
    const int nBlock = blockIdx.x * BN;

    // X loader: per instruction, a wave covers 4 rows x 32 cols (4x128B
    // contiguous segments -> fully used cachelines).
    const int rbase = tid >> 3;          // 0..31
    const int cb    = (tid & 7) * 4;     // 0,4,...,28
    // Weff loader: thread covers 16 bf16 (32B) of one row.
    const int wrow  = tid >> 1;          // 0..127
    const int wcol  = (tid & 1) * 16;    // 0 or 16

    // uniform bases (SGPR) + divergent 32-bit element offsets (VGPR)
    const float*  xbase = X    + (size_t)mBlock * KD;   // uniform
    const __bf16* wbase = Weff + (size_t)nBlock * KD;   // uniform
    const int xoff = rbase * KD + cb;                   // divergent
    const int woff = wrow  * KD + wcol;                 // divergent

    v8f acc[4][4];
    const v8f vzero = (v8f)(0.0f);
#pragma unroll
    for (int i = 0; i < 4; ++i)
#pragma unroll
        for (int j = 0; j < 4; ++j) acc[i][j] = vzero;

    auto load_tile = [&](int kt, int buf) {
        // --- X tile: 8 saddr-form float4 loads (imm row offsets when KK>0),
        //     cvt_pk to bf16, b64 stores to LDS ---
        const float* xs = xbase + kt;    // uniform part
        float4 xv[8];
#pragma unroll
        for (int e = 0; e < 8; ++e)
            xv[e] = *(const float4*)(xs + xoff + e * 32 * KD);
#pragma unroll
        for (int e = 0; e < 8; ++e) {
            v2bf lo = pack2bf(xv[e].x, xv[e].y);
            v2bf hi = pack2bf(xv[e].z, xv[e].w);
            v4bf c;
            c[0] = lo[0]; c[1] = lo[1]; c[2] = hi[0]; c[3] = hi[1];
            *(v4bf*)&As[buf][(rbase + e * 32) * LDSS + cb] = c;
        }
        // --- Weff tile: already bf16 -> async DMA straight into LDS.
        //     INST_OFFSET applies to both LDS and global addresses, so two
        //     b128 ops move this thread's 32 B. Tracked by ASYNCcnt. ---
        const __bf16* g = wbase + (woff + kt);
        unsigned ldsaddr = (unsigned)(size_t)&Bs[buf][wrow * LDSS + wcol];
        asm volatile(
            "global_load_async_to_lds_b128 %0, %1, off\n\t"
            "global_load_async_to_lds_b128 %0, %1, off offset:16"
            :: "v"(ldsaddr), "v"(g) : "memory");
        // --- prefetch X two K-tiles ahead (global_prefetch_b8) ---
        if (kt + 2 * BK < KD) {
            __builtin_prefetch(xs + xoff + 2 * BK, 0, 3);
        }
    };

    auto compute_tile = [&](int buf) {
        // A-fragment (16x32 bf16, ISA 7.12.2): lanes 0-15 row M=l16,
        // elems 0-7 = K0..7, elems 8-15 = K16..23; lanes 16-31 K+8.
        v16bf af[4];
#pragma unroll
        for (int i = 0; i < 4; ++i) {
            const __bf16* p = &As[buf][(wm * 64 + i * 16 + l16) * LDSS];
            v8bf c0 = *(const v8bf*)(p + halfSel * 8);
            v8bf c1 = *(const v8bf*)(p + 16 + halfSel * 8);
#pragma unroll
            for (int e = 0; e < 8; ++e) { af[i][e] = c0[e]; af[i][e + 8] = c1[e]; }
        }
        // B-fragment (32x16 bf16): col N=l16, same K order as A:
        // elems 0-7 = K0..7, elems 8-15 = K16..23; lanes 16-31 K+8.
        v16bf bfr[4];
#pragma unroll
        for (int j = 0; j < 4; ++j) {
            const __bf16* p = &Bs[buf][(wn * 64 + j * 16 + l16) * LDSS];
            v8bf c0 = *(const v8bf*)(p + halfSel * 8);
            v8bf c1 = *(const v8bf*)(p + 16 + halfSel * 8);
#pragma unroll
            for (int e = 0; e < 8; ++e) { bfr[j][e] = c0[e]; bfr[j][e + 8] = c1[e]; }
        }
#pragma unroll
        for (int i = 0; i < 4; ++i)
#pragma unroll
            for (int j = 0; j < 4; ++j)
                acc[i][j] = __builtin_amdgcn_wmma_f32_16x16x32_bf16(
                    false, af[i], false, bfr[j],
                    (short)0, acc[i][j], false, false);
    };

    // double-buffered main loop over K; async LDS writes fenced by
    // s_wait_asynccnt before every barrier.
    load_tile(0, 0);
    asm volatile("s_wait_asynccnt 0x0" ::: "memory");
    __syncthreads();
    const int numK = KD / BK;
    for (int t = 0; t < numK; ++t) {
        const int buf = t & 1;
        if (t + 1 < numK) load_tile((t + 1) * BK, buf ^ 1);
        compute_tile(buf);
        asm volatile("s_wait_asynccnt 0x0" ::: "memory");
        __syncthreads();
    }

    // epilogue: C/D layout -> VGPR v holds row (v + 8*halfSel), col l16
#pragma unroll
    for (int j = 0; j < 4; ++j) {
        const int col = nBlock + wn * 64 + j * 16 + l16;
        const float bv = bias[col];
#pragma unroll
        for (int i = 0; i < 4; ++i) {
            const int rowBase = mBlock + wm * 64 + i * 16 + halfSel * 8;
#pragma unroll
            for (int v = 0; v < 8; ++v) {
                Out[(size_t)(rowBase + v) * N + col] = acc[i][j][v] + bv;
            }
        }
    }
}

// ---------------------------------------------------------------------------
// Host launcher
// ---------------------------------------------------------------------------
extern "C" void kernel_launch(void* const* d_in, const int* in_sizes, int n_in,
                              void* d_out, int out_size, void* d_ws, size_t ws_size,
                              hipStream_t stream) {
    (void)n_in; (void)out_size; (void)ws_size;

    const float* x  = (const float*)d_in[0];   // [M, Din] flat ([B,S,Din])
    const float* W  = (const float*)d_in[1];   // [Dout, Din]
    const float* b  = (const float*)d_in[2];   // [Dout]
    const float* A  = (const float*)d_in[3];   // [r, Din]
    const float* Bm = (const float*)d_in[4];   // [Dout, r]
    float* out      = (float*)d_out;           // [M, Dout]

    const int Dout = in_sizes[2];
    const int Din  = in_sizes[1] / Dout;
    const int M    = in_sizes[0] / Din;
    const int r    = in_sizes[4] / Dout;
    const float scaling = 32.0f / 16.0f;       // lora_alpha / r

    __bf16* Weff = (__bf16*)d_ws;              // Dout*Din*2 bytes = 32 MB

    // 1) fold LoRA into the weight, cast to bf16
    dim3 blk1(256);
    dim3 grd1((Din / 4 + 255) / 256, Dout);
    fuse_weights_kernel<<<grd1, blk1, 0, stream>>>(W, A, Bm, Weff,
                                                   Dout, Din, r, scaling);

    // 2) single fused GEMM + bias (K specialized at compile time when 4096)
    dim3 blk2(256);
    dim3 grd2(Dout / BN, M / BM);
    if (Din == 4096) {
        lora_gemm_kernel<4096><<<grd2, blk2, 0, stream>>>(x, Weff, b, out,
                                                          M, Dout, Din);
    } else {
        lora_gemm_kernel<0><<<grd2, blk2, 0, stream>>>(x, Weff, b, out,
                                                       M, Dout, Din);
    }
}
